// DenseLayer_16793322127439
// MI455X (gfx1250) — matrix operand order repaired and checked by reading, not compiled, not run on hardware
//
#include <hip/hip_runtime.h>
#include <hip/hip_fp16.h>
#include <cmath>

typedef __attribute__((ext_vector_type(16))) _Float16 v16h;
typedef __attribute__((ext_vector_type(8)))  _Float16 v8h;
typedef __attribute__((ext_vector_type(8)))  float    v8f;

#define N_NODES 262144
#define N_EDGES 1048576
#define D_IN    384
#define D_BN    128
#define NBOND   4
#define D_AGG   512     /* NBOND * D_BN */
#define D_Z     640     /* (1+NBOND) * D_BN */
#define D_OUT   32
#define BN_EPS  1e-5f

// Branchless ELU using the hardware v_exp_f32 path.
// x>0:  max(x,0)=x,  exp(x)-1>0 so min(.,0)=0  -> x
// x<=0: max(x,0)=0,  exp(x)-1<=0               -> exp(x)-1
__device__ __forceinline__ float fast_elu(float x) {
    return fmaxf(x, 0.f) + fminf(__expf(x) - 1.f, 0.f);
}

// ---------------------------------------------------------------------------
// Weight pre-transpose + f16 conversion:  Wt[n][k] = (f16) W[k][n]
// ---------------------------------------------------------------------------
__global__ void conv_w1_kernel(const float* __restrict__ W, _Float16* __restrict__ Wt) {
    int idx = blockIdx.x * 256 + threadIdx.x;          // over D_BN * D_IN
    if (idx >= D_BN * D_IN) return;
    int n = idx / D_IN;
    int k = idx - n * D_IN;
    Wt[idx] = (_Float16)W[k * D_BN + n];
}

__global__ void conv_w2_kernel(const float* __restrict__ W, _Float16* __restrict__ Wt) {
    int idx = blockIdx.x * 256 + threadIdx.x;          // over D_OUT * D_Z
    if (idx >= D_OUT * D_Z) return;
    int n = idx / D_Z;
    int k = idx - n * D_Z;
    Wt[idx] = (_Float16)W[k * D_OUT + n];
}

// ---------------------------------------------------------------------------
// Zero the f16 aggregation buffer (ws is not re-poisoned between replays)
// ---------------------------------------------------------------------------
__global__ void zero_kernel(float4* __restrict__ p, long n4) {
    long i = (long)blockIdx.x * blockDim.x + threadIdx.x;
    long stride = (long)gridDim.x * blockDim.x;
    float4 z; z.x = 0.f; z.y = 0.f; z.z = 0.f; z.w = 0.f;
    for (; i < n4; i += stride) p[i] = z;
}

// ---------------------------------------------------------------------------
// Fused bottleneck: h = ELU(BN(concat(x1,x2,x3))) @ W1   via v_wmma f16
// Block = 256 threads = 8 waves; each wave owns 16 rows x 128 cols.
// h is stored in f16 (matches downstream WMMA input precision).
// ---------------------------------------------------------------------------
__global__ __launch_bounds__(256) void fused_bottleneck_kernel(
    const float* __restrict__ x1, const float* __restrict__ x2,
    const float* __restrict__ x3,
    const float* __restrict__ gamma, const float* __restrict__ beta,
    const float* __restrict__ mean,  const float* __restrict__ var,
    const _Float16* __restrict__ Wt, _Float16* __restrict__ h)
{
    __shared__ float s_scale[D_IN];
    __shared__ float s_shift[D_IN];
    const int tid = threadIdx.x;
    for (int c = tid; c < D_IN; c += 256) {
        float sc = gamma[c] * rsqrtf(var[c] + BN_EPS);
        s_scale[c] = sc;
        s_shift[c] = beta[c] - mean[c] * sc;
    }
    __syncthreads();

    const int wave = tid >> 5;
    const int lane = tid & 31;
    const int m    = lane & 15;          // A row / B,D column within tile
    const int half = lane >> 4;
    const long rowA = (long)blockIdx.x * 128 + wave * 16 + m;

    v8f acc[8];
    const v8f vzero = {0.f, 0.f, 0.f, 0.f, 0.f, 0.f, 0.f, 0.f};
#pragma unroll
    for (int nt = 0; nt < 8; ++nt) acc[nt] = vzero;

#pragma unroll
    for (int kc = 0; kc < 12; ++kc) {
        const int k0 = kc * 32 + half * 8;
        v16h a;
#pragma unroll
        for (int ch = 0; ch < 2; ++ch) {
            const int kk = k0 + ch * 16;   // compile-time constant source select
            const float* src = (kk < 128) ? (x1 + rowA * 128 + kk)
                             : (kk < 256) ? (x2 + rowA * 128 + (kk - 128))
                                          : (x3 + rowA * 128 + (kk - 256));
            float4 p0 = *(const float4*)(src);
            float4 p1 = *(const float4*)(src + 4);
            float v[8] = {p0.x, p0.y, p0.z, p0.w, p1.x, p1.y, p1.z, p1.w};
#pragma unroll
            for (int i = 0; i < 8; ++i) {
                float xn = v[i] * s_scale[kk + i] + s_shift[kk + i];
                a[ch * 8 + i] = (_Float16)fast_elu(xn);
            }
        }
        const int kb = kc * 32 + half * 8;    // B fragment: k = kb + e (e<8), kb + 16 + (e-8) (e>=8)
#pragma unroll
        for (int nt = 0; nt < 8; ++nt) {
            const _Float16* bp = Wt + (nt * 16 + m) * D_IN + kb;
            v8h blo = *(const v8h*)(bp);
            v8h bhi = *(const v8h*)(bp + 16);
            v16h b = __builtin_shufflevector(blo, bhi,
                     0,1,2,3,4,5,6,7,8,9,10,11,12,13,14,15);
            acc[nt] = __builtin_amdgcn_wmma_f32_16x16x32_f16(
                false, a, false, b, (short)0, acc[nt], false, false);
        }
    }

    const long rowD = (long)blockIdx.x * 128 + wave * 16 + half * 8;
#pragma unroll
    for (int nt = 0; nt < 8; ++nt) {
        const int col = nt * 16 + m;
#pragma unroll
        for (int r = 0; r < 8; ++r)
            h[(rowD + r) * D_BN + col] = (_Float16)acc[nt][r];
    }
}

// ---------------------------------------------------------------------------
// Edge scatter: agg[begin*4+bond][:] += h[end][:]
// f16 rows, packed-f16 hardware atomics (global_atomic_pk_add_f16).
// 64 threads per edge (2 features each), 4 edges per 256-thread block.
// ---------------------------------------------------------------------------
__global__ __launch_bounds__(256) void scatter_kernel(
    const _Float16* __restrict__ h,
    const int* __restrict__ begin_ids, const int* __restrict__ end_ids,
    const int* __restrict__ bond_types, _Float16* __restrict__ agg)
{
    const int t = threadIdx.x;
    int e = (int)blockIdx.x * 4 + (t >> 6);   // wave-uniform
    e = __builtin_amdgcn_readfirstlane(e);
    const int f = t & 63;                      // __half2 index (2 feats)
    const int nb = __builtin_amdgcn_readfirstlane(begin_ids[e]);
    const int ne = __builtin_amdgcn_readfirstlane(end_ids[e]);
    const int bt = __builtin_amdgcn_readfirstlane(bond_types[e]);

    const __half2* hv = (const __half2*)(h + (long)ne * D_BN);
    __half2 v = hv[f];
    __half2* dst = (__half2*)(agg + ((long)nb * NBOND + bt) * D_BN);
    atomicAdd(dst + f, v);
}

// ---------------------------------------------------------------------------
// Fused output: out = ELU(BN(concat(h,agg))) @ W2   via v_wmma f16
// Block = 256 threads = 8 waves; each wave owns 16 rows x 32 cols.
// ---------------------------------------------------------------------------
__global__ __launch_bounds__(256) void fused_output_kernel(
    const _Float16* __restrict__ h, const _Float16* __restrict__ agg,
    const float* __restrict__ gamma, const float* __restrict__ beta,
    const float* __restrict__ mean,  const float* __restrict__ var,
    const _Float16* __restrict__ Wt, float* __restrict__ out)
{
    __shared__ float s_scale[D_Z];
    __shared__ float s_shift[D_Z];
    const int tid = threadIdx.x;
    for (int c = tid; c < D_Z; c += 256) {
        float sc = gamma[c] * rsqrtf(var[c] + BN_EPS);
        s_scale[c] = sc;
        s_shift[c] = beta[c] - mean[c] * sc;
    }
    __syncthreads();

    const int wave = tid >> 5;
    const int lane = tid & 31;
    const int m    = lane & 15;
    const int half = lane >> 4;
    const long rowA = (long)blockIdx.x * 128 + wave * 16 + m;

    v8f acc[2];
    const v8f vzero = {0.f, 0.f, 0.f, 0.f, 0.f, 0.f, 0.f, 0.f};
    acc[0] = vzero; acc[1] = vzero;

#pragma unroll
    for (int kc = 0; kc < 20; ++kc) {
        const int k0 = kc * 32 + half * 8;
        v16h a;
#pragma unroll
        for (int ch = 0; ch < 2; ++ch) {
            const int kk = k0 + ch * 16;   // compile-time constant source select
            const _Float16* src = (kk < 128) ? (h + rowA * D_BN + kk)
                                             : (agg + rowA * D_AGG + (kk - 128));
            v8h zin = *(const v8h*)(src);
#pragma unroll
            for (int i = 0; i < 8; ++i) {
                float xn = (float)zin[i] * s_scale[kk + i] + s_shift[kk + i];
                a[ch * 8 + i] = (_Float16)fast_elu(xn);
            }
        }
        const int kb = kc * 32 + half * 8;
#pragma unroll
        for (int nt = 0; nt < 2; ++nt) {
            const _Float16* bp = Wt + (nt * 16 + m) * D_Z + kb;
            v8h blo = *(const v8h*)(bp);
            v8h bhi = *(const v8h*)(bp + 16);
            v16h b = __builtin_shufflevector(blo, bhi,
                     0,1,2,3,4,5,6,7,8,9,10,11,12,13,14,15);
            acc[nt] = __builtin_amdgcn_wmma_f32_16x16x32_f16(
                false, a, false, b, (short)0, acc[nt], false, false);
        }
    }

    const long rowD = (long)blockIdx.x * 128 + wave * 16 + half * 8;
#pragma unroll
    for (int nt = 0; nt < 2; ++nt) {
        const int col = nt * 16 + m;
#pragma unroll
        for (int r = 0; r < 8; ++r)
            out[(rowD + r) * D_OUT + col] = acc[nt][r];
    }
}

// ---------------------------------------------------------------------------
extern "C" void kernel_launch(void* const* d_in, const int* in_sizes, int n_in,
                              void* d_out, int out_size, void* d_ws, size_t ws_size,
                              hipStream_t stream) {
    const float* x1 = (const float*)d_in[0];
    const float* x2 = (const float*)d_in[1];
    const float* x3 = (const float*)d_in[2];
    const int* begin_ids  = (const int*)d_in[3];
    const int* end_ids    = (const int*)d_in[4];
    const int* bond_types = (const int*)d_in[5];
    const float* bn1_gamma = (const float*)d_in[6];
    const float* bn1_beta  = (const float*)d_in[7];
    const float* bn1_mean  = (const float*)d_in[8];
    const float* bn1_var   = (const float*)d_in[9];
    const float* W1        = (const float*)d_in[10];
    const float* bn2_gamma = (const float*)d_in[11];
    const float* bn2_beta  = (const float*)d_in[12];
    const float* bn2_mean  = (const float*)d_in[13];
    const float* bn2_var   = (const float*)d_in[14];
    const float* W2        = (const float*)d_in[15];
    float* out = (float*)d_out;

    char* w = (char*)d_ws;
    const size_t h_bytes   = (size_t)N_NODES * D_BN * sizeof(_Float16);  //  64 MB
    const size_t agg_bytes = (size_t)N_NODES * D_AGG * sizeof(_Float16); // 256 MB
    _Float16*  h_buf   = (_Float16*)w;
    _Float16*  agg_buf = (_Float16*)(w + h_bytes);
    _Float16*  W1t     = (_Float16*)(w + h_bytes + agg_bytes);
    _Float16*  W2t     = W1t + (size_t)D_BN * D_IN;

    conv_w1_kernel<<<(D_BN * D_IN + 255) / 256, 256, 0, stream>>>(W1, W1t);
    conv_w2_kernel<<<(D_OUT * D_Z + 255) / 256, 256, 0, stream>>>(W2, W2t);
    zero_kernel<<<8192, 256, 0, stream>>>((float4*)agg_buf,
                                          (long)(agg_bytes / sizeof(float4)));

    fused_bottleneck_kernel<<<N_NODES / 128, 256, 0, stream>>>(
        x1, x2, x3, bn1_gamma, bn1_beta, bn1_mean, bn1_var, W1t, h_buf);

    scatter_kernel<<<N_EDGES / 4, 256, 0, stream>>>(
        h_buf, begin_ids, end_ids, bond_types, agg_buf);

    fused_output_kernel<<<N_NODES / 128, 256, 0, stream>>>(
        h_buf, agg_buf, bn2_gamma, bn2_beta, bn2_mean, bn2_var, W2t, out);
}
